// Attention_24704651886606
// MI455X (gfx1250) — matrix operand order repaired and checked by reading, not compiled, not run on hardware
//
#include <hip/hip_runtime.h>
#include <hip/hip_bf16.h>
#include <math.h>

#define B_DIM 64
#define T_DIM 2048
#define ENC   1024
#define DEC   1024
#define A_DIM 512

typedef __attribute__((ext_vector_type(16))) __bf16 v16bf;
typedef __attribute__((ext_vector_type(8)))  __bf16 v8bf;
typedef __attribute__((ext_vector_type(8)))  float  v8f;

__device__ __forceinline__ unsigned short f32_to_bf16(float f) {
    unsigned u = __builtin_bit_cast(unsigned, f);
    u += 0x7FFFu + ((u >> 16) & 1u);          // round-to-nearest-even
    return (unsigned short)(u >> 16);
}

#if __has_builtin(__builtin_amdgcn_tanhf)
__device__ __forceinline__ float fast_tanh(float x) { return __builtin_amdgcn_tanhf(x); }
#elif __has_builtin(__builtin_amdgcn_tanh_f32)
__device__ __forceinline__ float fast_tanh(float x) { return __builtin_amdgcn_tanh_f32(x); }
#else
__device__ __forceinline__ float fast_tanh(float x) { return tanhf(x); }
#endif

// ---------------------------------------------------------------------------
// Kernel 1: f2[b,a] = sum_e s_t[b,e] * W_w[e,a]   (tiny: 64x1024x512)
// ---------------------------------------------------------------------------
__global__ void f2_kernel(const float* __restrict__ s_t,
                          const float* __restrict__ W_w,
                          float* __restrict__ f2) {
    const int b = blockIdx.x;
    const int a = threadIdx.x;                 // 0..511
    const float* srow = s_t + (size_t)b * DEC;
    float acc = 0.f;
    for (int e = 0; e < DEC; ++e)
        acc += srow[e] * W_w[(size_t)e * A_DIM + a];
    f2[(size_t)b * A_DIM + a] = acc;
}

// ---------------------------------------------------------------------------
// Kernel 2: U_t[a][k] = bf16(U_w[k][a])  (transpose + convert, 1 MB output)
// ---------------------------------------------------------------------------
__global__ void ut_kernel(const float* __restrict__ U_w,
                          unsigned short* __restrict__ U_t) {
    const int idx = blockIdx.x * 256 + threadIdx.x;   // a*1024 + k
    const int a = idx >> 10;
    const int k = idx & 1023;
    U_t[idx] = f32_to_bf16(U_w[(size_t)k * A_DIM + a]);
}

// ---------------------------------------------------------------------------
// Kernel 3: scores. Per block: 64-row M-tile of one batch; 8 waves split N=512.
// e[b,t] = sum_a tanh( (h_j@U)[t,a] + f2[b,a] ) * V[a]
// Double-buffered LDS A-tile: one barrier per K-step, next-tile global loads
// issued before the WMMAs that consume the current tile.
// ---------------------------------------------------------------------------
__global__ __launch_bounds__(256) void scores_kernel(
        const float* __restrict__ h_j,
        const unsigned short* __restrict__ U_t,
        const float* __restrict__ f2,
        const float* __restrict__ V_w,
        float* __restrict__ e_out) {
    __shared__ alignas(16) unsigned short lds_a[2][64 * 32]; // 2 x (64 rows x 32 k) bf16
    __shared__ float eacc[64];

    const int tid  = threadIdx.x;
    const int lane = tid & 31;
    const int wave = tid >> 5;
    const int b    = blockIdx.x >> 5;                     // 32 t-blocks per batch
    const int t0   = (blockIdx.x & 31) * 64;

    const size_t hbase = ((size_t)b * T_DIM + t0) * ENC;

    // cooperative A-tile staging: each thread converts 8 consecutive k of one row
    const int arow = tid >> 2;                 // 0..63
    const int akc  = (tid & 3) * 8;            // 0,8,16,24
    const float* hrow = h_j + hbase + (size_t)arow * ENC + akc;
    unsigned* lds_a32 = (unsigned*)lds_a;
    const int ldst = (arow * 32 + akc) >> 1;   // uint index within one buffer

    // fragment addressing (documented 16-bit WMMA layouts, wave32)
    const int half  = (lane < 16) ? 0 : 1;
    const int frow  = lane & 15;
    const int akoff = half ? 8 : 0;            // A: chunks at {akoff, akoff+16}
    const int bkoff = half ? 8 : 0;            // B: chunks at {bkoff, bkoff+16}
    const int c0    = wave * 64;               // this wave's N-slice

    v8f acc[4][4];
    #pragma unroll
    for (int m = 0; m < 4; ++m)
        #pragma unroll
        for (int n = 0; n < 4; ++n)
            acc[m][n] = (v8f){0.f,0.f,0.f,0.f,0.f,0.f,0.f,0.f};

    // prefetch tile 0
    float4 p0 = *(const float4*)(hrow);
    float4 p1 = *(const float4*)(hrow + 4);

    for (int kidx = 0; kidx < 32; ++kidx) {
        const int k0  = kidx * 32;
        const int buf = kidx & 1;

        // stage current tile into LDS buffer `buf`
        unsigned q0 = (unsigned)f32_to_bf16(p0.x) | ((unsigned)f32_to_bf16(p0.y) << 16);
        unsigned q1 = (unsigned)f32_to_bf16(p0.z) | ((unsigned)f32_to_bf16(p0.w) << 16);
        unsigned q2 = (unsigned)f32_to_bf16(p1.x) | ((unsigned)f32_to_bf16(p1.y) << 16);
        unsigned q3 = (unsigned)f32_to_bf16(p1.z) | ((unsigned)f32_to_bf16(p1.w) << 16);
        *(uint4*)&lds_a32[buf * 1024 + ldst] = make_uint4(q0, q1, q2, q3);

        // issue next tile's global loads early (hidden under barrier + WMMAs)
        if (kidx < 31) {
            p0 = *(const float4*)(hrow + k0 + 32);
            p1 = *(const float4*)(hrow + k0 + 36);
        }

        __syncthreads();   // tile `buf` visible; prior reads of buf finished

        // A fragments for the 4 m-subtiles (from LDS)
        v16bf af[4];
        #pragma unroll
        for (int m = 0; m < 4; ++m) {
            const unsigned short* base = &lds_a[buf][(m * 16 + frow) * 32 + akoff];
            v8bf lo = *(const v8bf*)base;
            v8bf hi = *(const v8bf*)(base + 16);
            af[m] = __builtin_shufflevector(lo, hi,
                     0,1,2,3,4,5,6,7,8,9,10,11,12,13,14,15);
        }
        // 4 n-subtiles; each B fragment reused by 4 WMMAs
        #pragma unroll
        for (int n = 0; n < 4; ++n) {
            const unsigned short* bbase =
                U_t + (size_t)(c0 + n * 16 + frow) * ENC + k0 + bkoff;
            v8bf lo = *(const v8bf*)bbase;
            v8bf hi = *(const v8bf*)(bbase + 16);
            v16bf bfrag = __builtin_shufflevector(lo, hi,
                     0,1,2,3,4,5,6,7,8,9,10,11,12,13,14,15);
            #pragma unroll
            for (int m = 0; m < 4; ++m)
                acc[m][n] = __builtin_amdgcn_wmma_f32_16x16x32_bf16(
                    false, af[m], false, bfrag, (short)0, acc[m][n], false, false);
        }
    }

    // epilogue: e[t] = sum_cols tanh(acc + f2[col]) * V[col]
    __syncthreads();
    if (tid < 64) eacc[tid] = 0.f;
    __syncthreads();

    float f2c[4], vc[4];
    #pragma unroll
    for (int n = 0; n < 4; ++n) {
        const int col = c0 + n * 16 + frow;
        f2c[n] = f2[(size_t)b * A_DIM + col];
        vc[n]  = V_w[col];
    }

    #pragma unroll
    for (int m = 0; m < 4; ++m) {
        float s[8];
        #pragma unroll
        for (int r = 0; r < 8; ++r) {
            float v = 0.f;
            #pragma unroll
            for (int n = 0; n < 4; ++n)
                v += fast_tanh(acc[m][n][r] + f2c[n]) * vc[n];
            // reduce across the 16 lanes of this half (cols)
            v += __shfl_xor(v, 1);
            v += __shfl_xor(v, 2);
            v += __shfl_xor(v, 4);
            v += __shfl_xor(v, 8);
            s[r] = v;
        }
        if (frow == 0) {
            #pragma unroll
            for (int r = 0; r < 8; ++r)
                atomicAdd(&eacc[m * 16 + half * 8 + r], s[r]);
        }
    }
    __syncthreads();
    if (tid < 64)
        e_out[(size_t)b * T_DIM + t0 + tid] = eacc[tid];
}

// ---------------------------------------------------------------------------
// Kernel 4: softmax over T per batch
// ---------------------------------------------------------------------------
__global__ void softmax_kernel(const float* __restrict__ e_in,
                               float* __restrict__ a_out) {
    __shared__ float red[256];
    const int b = blockIdx.x;
    const int tid = threadIdx.x;
    const float* e = e_in + (size_t)b * T_DIM;

    float vals[8];
    float m = -3.402823466e38f;
    #pragma unroll
    for (int i = 0; i < 8; ++i) {
        vals[i] = e[tid + i * 256];
        m = fmaxf(m, vals[i]);
    }
    red[tid] = m; __syncthreads();
    for (int s = 128; s > 0; s >>= 1) {
        if (tid < s) red[tid] = fmaxf(red[tid], red[tid + s]);
        __syncthreads();
    }
    m = red[0]; __syncthreads();

    float sum = 0.f;
    #pragma unroll
    for (int i = 0; i < 8; ++i) {
        vals[i] = __expf(vals[i] - m);
        sum += vals[i];
    }
    red[tid] = sum; __syncthreads();
    for (int s = 128; s > 0; s >>= 1) {
        if (tid < s) red[tid] += red[tid + s];
        __syncthreads();
    }
    const float inv = 1.f / red[0];
    #pragma unroll
    for (int i = 0; i < 8; ++i)
        a_out[(size_t)b * T_DIM + tid + i * 256] = vals[i] * inv;
}

// ---------------------------------------------------------------------------
// Kernel 5: c[b,:] = sum_t a[b,t] * h_j[b,t,:]   (streams h_j once)
// ---------------------------------------------------------------------------
__global__ void context_kernel(const float* __restrict__ h_j,
                               const float* __restrict__ a_t,
                               float* __restrict__ c_out) {
    const int b = blockIdx.x;
    const int tid = threadIdx.x;               // 0..255, owns 4 cols via float4
    const float4* hb = (const float4*)(h_j + (size_t)b * T_DIM * ENC);
    const float* ab = a_t + (size_t)b * T_DIM;
    float4 acc = make_float4(0.f, 0.f, 0.f, 0.f);
    for (int t = 0; t < T_DIM; ++t) {
        const float a = ab[t];
        const float4 h = hb[(size_t)t * (ENC / 4) + tid];
        acc.x += a * h.x; acc.y += a * h.y; acc.z += a * h.z; acc.w += a * h.w;
    }
    ((float4*)c_out)[(size_t)b * (ENC / 4) + tid] = acc;
}

// ---------------------------------------------------------------------------
extern "C" void kernel_launch(void* const* d_in, const int* in_sizes, int n_in,
                              void* d_out, int out_size, void* d_ws, size_t ws_size,
                              hipStream_t stream) {
    const float* h_j = (const float*)d_in[0];
    const float* s_t = (const float*)d_in[1];
    const float* U_w = (const float*)d_in[2];
    const float* W_w = (const float*)d_in[3];
    const float* V_w = (const float*)d_in[4];

    float* out   = (float*)d_out;
    float* c_out = out;                          // [64,1,1024] first
    float* a_out = out + (size_t)B_DIM * ENC;    // [64,2048,1] second

    // workspace layout
    float* f2  = (float*)d_ws;                                   // 64*512 f32
    float* e_s = f2 + (size_t)B_DIM * A_DIM;                     // 64*2048 f32
    unsigned short* U_t = (unsigned short*)(e_s + (size_t)B_DIM * T_DIM); // 512*1024 bf16

    f2_kernel     <<<B_DIM, 512, 0, stream>>>(s_t, W_w, f2);
    ut_kernel     <<<(A_DIM * ENC) / 256, 256, 0, stream>>>(U_w, U_t);
    scores_kernel <<<B_DIM * (T_DIM / 64), 256, 0, stream>>>(h_j, U_t, f2, V_w, e_s);
    softmax_kernel<<<B_DIM, 256, 0, stream>>>(e_s, a_out);
    context_kernel<<<B_DIM, 256, 0, stream>>>(h_j, a_out, c_out);
}
